// GINEncoder_65111704207428
// MI455X (gfx1250) — compile-verified, matrix operand order checked by reading, not run on hardware
//
#include <hip/hip_runtime.h>
#include <hip/hip_bf16.h>

typedef __attribute__((ext_vector_type(16))) __bf16 v16bf;
typedef __attribute__((ext_vector_type(8)))  float  v8f;
typedef int v4i __attribute__((vector_size(16)));

union WOp { v16bf v; unsigned u[8]; };

#define GAS __attribute__((address_space(1)))
#define LAS __attribute__((address_space(3)))

#if __has_builtin(__builtin_amdgcn_global_load_async_to_lds_b128)
#define HAVE_ASYNC_LDS 1
#else
#define HAVE_ASYNC_LDS 0
#endif

static __device__ inline unsigned short f2bf(float f) {
    unsigned u = __float_as_uint(f);
    unsigned r = (u + 0x7FFFu + ((u >> 16) & 1u)) >> 16;
    return (unsigned short)r;
}

static __device__ inline void wait_async_lds() {
#if HAVE_ASYNC_LDS
#if __has_builtin(__builtin_amdgcn_s_wait_asynccnt)
    __builtin_amdgcn_s_wait_asynccnt(0);
#else
    asm volatile("s_wait_asynccnt 0x0" ::: "memory");
#endif
#endif
}

// ------------- one-time weight convert: f32 -> bf16 in LDS-ready layout ------
// cW1[l][c*64+k] = bf16(W1[l][k][c]),  cW2[l][c*128+k] = bf16(W2[l][k][c])
__global__ void gin_cvt_w(const float* __restrict__ W1, const float* __restrict__ W2,
                          unsigned short* __restrict__ cW1, unsigned short* __restrict__ cW2,
                          int L) {
    int i = blockIdx.x * blockDim.x + threadIdx.x;
    int total = L * 8192;
    if (i < total) {
        int l = i >> 13, r = i & 8191;
        int c = r >> 6, k = r & 63;
        cW1[i] = f2bf(W1[(size_t)l * 8192 + k * 128 + c]);
    } else if (i < 2 * total) {
        int j = i - total;
        int l = j >> 13, r = j & 8191;
        int c = r >> 7, k = r & 127;
        cW2[j] = f2bf(W2[(size_t)l * 8192 + k * 64 + c]);
    }
}

// ---------------- z = h (vectorized copy, float4 granularity) ----------------
__global__ void gin_init_z(float* __restrict__ z, const float* __restrict__ h, int n4) {
    int i = blockIdx.x * blockDim.x + threadIdx.x;
    if (i < n4) ((float4*)z)[i] = ((const float4*)h)[i];
}

// ---------------- z[dst] += h[src] over edges (16 threads / edge) ------------
__global__ void gin_scatter(float* __restrict__ z, const float* __restrict__ h,
                            const int* __restrict__ src, const int* __restrict__ dst, int E) {
    int t = blockIdx.x * blockDim.x + threadIdx.x;
    int e = t >> 4;
    if (e >= E) return;
    int c = (t & 15) * 4;
    int s = src[e], d = dst[e];
    float4 val = *(const float4*)(h + s * 64 + c);
    float* p = z + d * 64 + c;
    atomicAdd(p + 0, val.x);
    atomicAdd(p + 1, val.y);
    atomicAdd(p + 2, val.z);
    atomicAdd(p + 3, val.w);
}

// ---------------- fused MLP: z = BN2(W2ᵀ·relu(BN1(W1ᵀ·z)))  (in-place) -------
// Block = 128 threads = 4 waves; each wave does one 16-row WMMA tile.
__global__ __launch_bounds__(128) void gin_mlp(
    float* __restrict__ z, int N,
    const unsigned short* __restrict__ cW1,   // [128*64]  bf16, layout c*64+k
    const unsigned short* __restrict__ cW2,   // [64*128]  bf16, layout c*128+k
    const float* __restrict__ b1, const float* __restrict__ g1,
    const float* __restrict__ bt1, const float* __restrict__ m1,
    const float* __restrict__ v1,
    const float* __restrict__ b2, const float* __restrict__ g2,
    const float* __restrict__ bt2, const float* __restrict__ m2,
    const float* __restrict__ v2,
    int relu_out)
{
    __shared__ __align__(16) unsigned short sW1[128 * 64];    // 16KB
    __shared__ __align__(16) unsigned short sW2[64 * 128];    // 16KB
    __shared__ float sS1[128], sT1[128], sS2[64], sT2[64];
    __shared__ __align__(16) unsigned short sA[4][16 * 64];   //  8KB
    __shared__ __align__(16) unsigned short sZ1[4][16 * 128]; // 16KB

    const int tid = threadIdx.x;

    // ---- stage bf16 weights into LDS (async DMA if available) ----
#if HAVE_ASYNC_LDS
    {
        GAS v4i* gw1 = (GAS v4i*)cW1;    // 8 bf16 per 16B transfer
        GAS v4i* gw2 = (GAS v4i*)cW2;
        LAS v4i* lw1 = (LAS v4i*)sW1;
        LAS v4i* lw2 = (LAS v4i*)sW2;
        for (int i = tid; i < 1024; i += 128) {
            __builtin_amdgcn_global_load_async_to_lds_b128(gw1 + i, lw1 + i, 0, 0);
            __builtin_amdgcn_global_load_async_to_lds_b128(gw2 + i, lw2 + i, 0, 0);
        }
    }
#else
    for (int i = tid; i < 1024; i += 128) {
        ((uint4*)sW1)[i] = ((const uint4*)cW1)[i];
        ((uint4*)sW2)[i] = ((const uint4*)cW2)[i];
    }
#endif

    // ---- folded BN affines (overlaps with async weight DMA) ----
    if (tid < 128) {
        float s = g1[tid] * rsqrtf(v1[tid] + 1e-5f);
        sS1[tid] = s;
        sT1[tid] = (b1[tid] - m1[tid]) * s + bt1[tid];
    }
    if (tid < 64) {
        float s = g2[tid] * rsqrtf(v2[tid] + 1e-5f);
        sS2[tid] = s;
        sT2[tid] = (b2[tid] - m2[tid]) * s + bt2[tid];
    }

    const int wave = tid >> 5, lane = tid & 31;
    const int row0 = blockIdx.x * 64 + wave * 16;
    unsigned short* A  = sA[wave];
    unsigned short* Z1 = sZ1[wave];

    // ---- load this wave's 16x64 f32 tile -> bf16 LDS (overlaps DMA) ----
    {
        int r = lane >> 1, hh = (lane & 1) * 32;
        int grow = row0 + r; if (grow > N - 1) grow = N - 1;
        const float4* s4 = (const float4*)(z + grow * 64 + hh);
#pragma unroll
        for (int q = 0; q < 8; ++q) {
            float4 f = s4[q];
            int base = r * 64 + hh + q * 4;
            A[base + 0] = f2bf(f.x); A[base + 1] = f2bf(f.y);
            A[base + 2] = f2bf(f.z); A[base + 3] = f2bf(f.w);
        }
    }
    wait_async_lds();
    __syncthreads();

    const int mrow = lane & 15;       // M (A) / N (B,C,D) index for this lane
    const int half = lane >> 4;       // lane half selects K sub-block

    // A operands for GEMM1: two K=32 blocks of the 16x64 tile
    WOp a0, a1;
#pragma unroll
    for (int v = 0; v < 8; ++v) {
        int k0 = ((v >> 2) << 4) + half * 8 + ((v & 3) << 1);
        a0.u[v] = *(const unsigned*)(A + mrow * 64 + k0);
        a1.u[v] = *(const unsigned*)(A + mrow * 64 + 32 + k0);
    }

    // GEMM1: 16x64 @ 64x128 -> 16x128, + BN1 + ReLU, to LDS as bf16
#pragma unroll
    for (int j = 0; j < 8; ++j) {
        int c = j * 16 + mrow;
        const unsigned short* wc = sW1 + c * 64;
        WOp bk0, bk1;
#pragma unroll
        for (int v = 0; v < 8; ++v) {
            int k0 = ((v >> 2) << 4) + half * 8 + ((v & 3) << 1);
            bk0.u[v] = *(const unsigned*)(wc + k0);
            bk1.u[v] = *(const unsigned*)(wc + 32 + k0);
        }
        v8f acc = {};
        acc = __builtin_amdgcn_wmma_f32_16x16x32_bf16(false, a0.v, false, bk0.v,
                                                      (short)0, acc, false, false);
        acc = __builtin_amdgcn_wmma_f32_16x16x32_bf16(false, a1.v, false, bk1.v,
                                                      (short)0, acc, false, false);
        float s = sS1[c], t = sT1[c];
#pragma unroll
        for (int v = 0; v < 8; ++v) {
            int mm = half * 8 + v;
            float val = fmaxf(acc[v] * s + t, 0.0f);
            Z1[mm * 128 + c] = f2bf(val);
        }
    }
    __syncthreads();

    // A operands for GEMM2: four K=32 blocks of the 16x128 hidden tile
    WOp a2[4];
#pragma unroll
    for (int kb = 0; kb < 4; ++kb)
#pragma unroll
        for (int v = 0; v < 8; ++v) {
            int k0 = kb * 32 + ((v >> 2) << 4) + half * 8 + ((v & 3) << 1);
            a2[kb].u[v] = *(const unsigned*)(Z1 + mrow * 128 + k0);
        }

    // GEMM2: 16x128 @ 128x64 -> 16x64, + BN2 (+ReLU), store f32 in-place
#pragma unroll
    for (int j = 0; j < 4; ++j) {
        int c = j * 16 + mrow;
        const unsigned short* wc = sW2 + c * 128;
        v8f acc = {};
#pragma unroll
        for (int kb = 0; kb < 4; ++kb) {
            WOp bb;
#pragma unroll
            for (int v = 0; v < 8; ++v) {
                int k0 = kb * 32 + ((v >> 2) << 4) + half * 8 + ((v & 3) << 1);
                bb.u[v] = *(const unsigned*)(wc + k0);
            }
            acc = __builtin_amdgcn_wmma_f32_16x16x32_bf16(false, a2[kb].v, false, bb.v,
                                                          (short)0, acc, false, false);
        }
        float s = sS2[c], t = sT2[c];
#pragma unroll
        for (int v = 0; v < 8; ++v) {
            int mm = half * 8 + v;
            int grow = row0 + mm;
            float val = acc[v] * s + t;
            if (relu_out) val = fmaxf(val, 0.0f);
            if (grow < N) z[grow * 64 + c] = val;
        }
    }
}

// ---------------- global mean pool -------------------------------------------
__global__ void pool_zero(float* __restrict__ sums, float* __restrict__ cnts, int G) {
    int i = blockIdx.x * blockDim.x + threadIdx.x;
    if (i < G * 64) sums[i] = 0.0f;
    if (i < G) cnts[i] = 0.0f;
}

__global__ void pool_acc(const float* __restrict__ h, const int* __restrict__ batch,
                         float* __restrict__ sums, float* __restrict__ cnts, int N) {
    int t = blockIdx.x * blockDim.x + threadIdx.x;
    int node = t >> 4;
    if (node >= N) return;
    int c = (t & 15) * 4;
    int g = batch[node];
    float4 val = *(const float4*)(h + node * 64 + c);
    float* p = sums + g * 64 + c;
    atomicAdd(p + 0, val.x);
    atomicAdd(p + 1, val.y);
    atomicAdd(p + 2, val.z);
    atomicAdd(p + 3, val.w);
    if (c == 0) atomicAdd(cnts + g, 1.0f);
}

__global__ void pool_div(float* __restrict__ out, const float* __restrict__ cnts, int G) {
    int i = blockIdx.x * blockDim.x + threadIdx.x;
    if (i < G * 64) out[i] = out[i] / fmaxf(cnts[i >> 6], 1.0f);
}

extern "C" void kernel_launch(void* const* d_in, const int* in_sizes, int n_in,
                              void* d_out, int out_size, void* d_ws, size_t ws_size,
                              hipStream_t stream) {
    const float* x     = (const float*)d_in[0];
    const int*   ei    = (const int*)d_in[1];
    const int*   batch = (const int*)d_in[2];
    const float* W1 = (const float*)d_in[3];
    const float* b1 = (const float*)d_in[4];
    const float* g1 = (const float*)d_in[5];
    const float* bt1 = (const float*)d_in[6];
    const float* m1 = (const float*)d_in[7];
    const float* v1 = (const float*)d_in[8];
    const float* W2 = (const float*)d_in[9];
    const float* b2 = (const float*)d_in[10];
    const float* g2 = (const float*)d_in[11];
    const float* bt2 = (const float*)d_in[12];
    const float* m2 = (const float*)d_in[13];
    const float* v2 = (const float*)d_in[14];

    const int N = in_sizes[2];          // 100000
    const int E = in_sizes[1] / 2;      // 1600000
    const int G = out_size / 64;        // 512
    const int L = 4;

    float* buf0 = (float*)d_ws;
    float* buf1 = buf0 + (size_t)N * 64;
    float* cnts = buf1 + (size_t)N * 64;
    unsigned short* cW1 = (unsigned short*)(cnts + G);          // L*8192 bf16
    unsigned short* cW2 = cW1 + (size_t)L * 8192;               // L*8192 bf16

    const int* srcI = ei;
    const int* dstI = ei + E;

    // one-time per-call weight conversion (f32 -> bf16, WMMA-B layout)
    {
        int tot = 2 * L * 8192;
        gin_cvt_w<<<(tot + 255) / 256, 256, 0, stream>>>(W1, W2, cW1, cW2, L);
    }

    const float* hcur = x;
    float* zb[2] = { buf0, buf1 };

    for (int l = 0; l < L; ++l) {
        float* zl = zb[l & 1];
        int n4 = N * 16;  // float4 count
        gin_init_z<<<(n4 + 255) / 256, 256, 0, stream>>>(zl, hcur, n4);
        gin_scatter<<<(E * 16 + 255) / 256, 256, 0, stream>>>(zl, hcur, srcI, dstI, E);
        gin_mlp<<<(N + 63) / 64, 128, 0, stream>>>(
            zl, N,
            cW1 + (size_t)l * 8192, cW2 + (size_t)l * 8192,
            b1 + l * 128, g1 + l * 128, bt1 + l * 128, m1 + l * 128, v1 + l * 128,
            b2 + l * 64, g2 + l * 64, bt2 + l * 64, m2 + l * 64, v2 + l * 64,
            (l != L - 1) ? 1 : 0);
        hcur = zl;
    }

    float* out = (float*)d_out;
    pool_zero<<<(G * 64 + 255) / 256, 256, 0, stream>>>(out, cnts, G);
    pool_acc<<<(N * 16 + 255) / 256, 256, 0, stream>>>(hcur, batch, out, cnts, N);
    pool_div<<<(G * 64 + 255) / 256, 256, 0, stream>>>(out, cnts, G);
}
